// nojit_customLSTMClassifier_22488448761886
// MI455X (gfx1250) — matrix operand order repaired and checked by reading, not compiled, not run on hardware
//
#include <hip/hip_runtime.h>
#include <cstdint>
#include <cstddef>

// ---------------------------------------------------------------------------
// Types for CDNA5 WMMA (wave32, v_wmma_f32_16x16x32_bf16)
// ---------------------------------------------------------------------------
typedef __bf16 bf16_t;
typedef __attribute__((ext_vector_type(8)))  bf16_t v8bf;
typedef __attribute__((ext_vector_type(16))) bf16_t v16bf;
typedef __attribute__((ext_vector_type(8)))  float  v8f;

union V16U { v16bf v; v8bf h[2]; };

static __device__ __forceinline__ bf16_t f2bf(float f) {
  // round-to-nearest-even fp32 -> bf16 (bit manipulation; no __bf16 cvt dependency)
  unsigned u = __builtin_bit_cast(unsigned, f);
  u += 0x7FFFu + ((u >> 16) & 1u);
  unsigned short s = (unsigned short)(u >> 16);
  return __builtin_bit_cast(bf16_t, s);
}

#define BATCH 64
#define SEQ   256
#define INDIM 256
#define EDIM  512
#define HID   1024
#define G4    4096            // 4*HID
#define ROWS  (BATCH * SEQ)   // 16384

// ---------------------------------------------------------------------------
// WMMA GEMM:  C[M,N] = A[M,K](bf16,row-major) x BT[N,K](bf16,row-major)^T
//             accumulator initialized from optional bias[N] and/or
//             accin[(m)*ldacc + n] (fp32, e.g. precomputed x-gates slice).
// Epilogue: prelu==1 -> PReLU(alpha) then bf16 store to Cbf
//           prelu==0 -> fp32 store to Cf
// Grid: (M/16, N/256), block 128 threads (4 waves, each wave = 16x64 tile).
// M,N,K must be multiples of 16/256/32 (true for all uses here) so EXEC is
// all-ones for every WMMA (ISA requirement).
// ---------------------------------------------------------------------------
__global__ __launch_bounds__(128)
void wmma_gemm_bf16(const bf16_t* __restrict__ A, int lda,
                    const bf16_t* __restrict__ BT, int ldb,
                    const float* __restrict__ bias,
                    const float* __restrict__ accin, long long ldacc,
                    float* __restrict__ Cf, bf16_t* __restrict__ Cbf, int ldc,
                    const float* __restrict__ alphap,
                    int K, int prelu)
{
  const int lane = threadIdx.x & 31;
  const int wave = threadIdx.x >> 5;
  const int row0 = blockIdx.x * 16;
  const int col0 = blockIdx.y * 256 + wave * 64;

  const int nl   = lane & 15;         // column within 16-wide tile / A-row
  const int hi   = lane >> 4;         // lane half
  const int ka   = hi * 16;           // A frag: K sub-offset (0 or 16), contiguous 16
  const int kb   = hi * 16;           // B frag: K sub-offset (0 or 16), contiguous 16
  const int mrow = hi * 8;            // C frag: m = r + 8*hi

  v8f acc[4];
  #pragma unroll
  for (int j = 0; j < 4; ++j) {
    const int n = col0 + j * 16 + nl;
    #pragma unroll
    for (int r = 0; r < 8; ++r) {
      float v = 0.0f;
      if (bias)  v  = bias[n];
      if (accin) v += accin[(size_t)(row0 + mrow + r) * (size_t)ldacc + n];
      acc[j][r] = v;
    }
  }

  const bf16_t* arow = A + (size_t)(row0 + nl) * lda;
  for (int k0 = 0; k0 < K; k0 += 32) {
    // A fragment: lane holds row m=nl, K = {ka..ka+15}
    V16U a;
    a.h[0] = *(const v8bf*)(arow + k0 + ka);
    a.h[1] = *(const v8bf*)(arow + k0 + ka + 8);
    // B fragments: lane holds column n, 16 consecutive K starting at kb
    v16bf bfr[4];
    #pragma unroll
    for (int j = 0; j < 4; ++j) {
      const bf16_t* bp = BT + (size_t)(col0 + j * 16 + nl) * ldb + k0 + kb;
      bfr[j] = *(const v16bf*)(bp);
    }
    #pragma unroll
    for (int j = 0; j < 4; ++j) {
      acc[j] = __builtin_amdgcn_wmma_f32_16x16x32_bf16(
          /*neg_a=*/false, a.v, /*neg_b=*/false, bfr[j],
          /*c_mod=*/(short)0, acc[j], /*reuse_a=*/false, /*reuse_b=*/false);
    }
  }

  const float alpha = alphap ? alphap[0] : 0.0f;
  #pragma unroll
  for (int j = 0; j < 4; ++j) {
    const int n = col0 + j * 16 + nl;
    #pragma unroll
    for (int r = 0; r < 8; ++r) {
      const size_t m = (size_t)(row0 + mrow + r);
      float v = acc[j][r];
      if (prelu) {
        v = (v >= 0.0f) ? v : alpha * v;
        Cbf[m * (size_t)ldc + n] = f2bf(v);
      } else {
        Cf[m * (size_t)ldc + n] = v;
      }
    }
  }
}

// ---------------------------------------------------------------------------
// Helpers
// ---------------------------------------------------------------------------
__global__ void f32_to_bf16_k(const float* __restrict__ in, bf16_t* __restrict__ outp, size_t n) {
  size_t i = (size_t)blockIdx.x * blockDim.x + threadIdx.x;
  if (i < n) outp[i] = f2bf(in[i]);
}

// W[K,N] fp32 -> WT[N,K] bf16 (one-time transpose-convert so GEMM B-loads are contiguous)
__global__ void w_transpose_bf16_k(const float* __restrict__ W, bf16_t* __restrict__ WT, int K, int N) {
  size_t i = (size_t)blockIdx.x * blockDim.x + threadIdx.x;
  if (i >= (size_t)K * N) return;
  int k = (int)(i / N), n = (int)(i % N);
  WT[(size_t)n * K + k] = f2bf(W[i]);
}

__global__ void vec_add_k(const float* __restrict__ a, const float* __restrict__ b,
                          float* __restrict__ o, int n) {
  int i = blockIdx.x * blockDim.x + threadIdx.x;
  if (i < n) o[i] = a[i] + b[i];
}

// LSTM cell pointwise update. gbuf[b,4H] holds f,i,g,o pre-activations.
__global__ void lstm_update_k(const float* __restrict__ gbuf,
                              float* __restrict__ c, float* __restrict__ h,
                              bf16_t* __restrict__ hbf, bf16_t* __restrict__ ys, int t) {
  int idx = blockIdx.x * blockDim.x + threadIdx.x;   // 0 .. BATCH*HID-1
  if (idx >= BATCH * HID) return;
  int b = idx >> 10, j = idx & (HID - 1);
  const float* g = gbuf + (size_t)b * G4;
  float fg = 1.0f / (1.0f + __expf(-g[j]));
  float ig = 1.0f / (1.0f + __expf(-g[HID + j]));
  float gg = tanhf(g[2 * HID + j]);
  float og = 1.0f / (1.0f + __expf(-g[3 * HID + j]));
  float cn = fg * c[idx] + ig * gg;
  float hn = og * tanhf(cn);
  c[idx] = cn;
  h[idx] = hn;
  hbf[idx] = f2bf(hn);
  if (ys) ys[((size_t)b * SEQ + t) * HID + j] = f2bf(hn);
}

__global__ void classifier_k(const float* __restrict__ h, const float* __restrict__ Wc,
                             const float* __restrict__ bc, float* __restrict__ outp) {
  int idx = blockIdx.x * blockDim.x + threadIdx.x;
  if (idx >= BATCH * 10) return;
  int b = idx / 10, k = idx % 10;
  float s = bc[k];
  for (int j = 0; j < HID; ++j) s += h[(size_t)b * HID + j] * Wc[(size_t)j * 10 + k];
  outp[idx] = s;
}

// ---------------------------------------------------------------------------
// Host launcher
// ---------------------------------------------------------------------------
extern "C" void kernel_launch(void* const* d_in, const int* in_sizes, int n_in,
                              void* d_out, int out_size, void* d_ws, size_t ws_size,
                              hipStream_t stream) {
  const float* x     = (const float*)d_in[0];
  const float* h0    = (const float*)d_in[1];
  const float* c0    = (const float*)d_in[2];
  const float* W_enc = (const float*)d_in[3];
  const float* b_enc = (const float*)d_in[4];
  const float* alpha = (const float*)d_in[5];
  const float* Wx0   = (const float*)d_in[6];
  const float* bx0   = (const float*)d_in[7];
  const float* Wh0   = (const float*)d_in[8];
  const float* bh0   = (const float*)d_in[9];
  const float* Wx1   = (const float*)d_in[10];
  const float* bx1   = (const float*)d_in[11];
  const float* Wh1   = (const float*)d_in[12];
  const float* bh1   = (const float*)d_in[13];
  const float* Wcls  = (const float*)d_in[14];
  const float* bcls  = (const float*)d_in[15];
  float* out = (float*)d_out;

  char* ws = (char*)d_ws;
  size_t off = 0;
  auto alloc = [&](size_t bytes) -> void* {
    void* p = ws + off;
    off += (bytes + 255) & ~(size_t)255;
    return p;
  };
  bf16_t* xbf   = (bf16_t*)alloc((size_t)ROWS * INDIM * 2);   //   8 MB
  bf16_t* WencT = (bf16_t*)alloc((size_t)EDIM * INDIM * 2);   // 256 KB  [E, IN]
  bf16_t* Wx0T  = (bf16_t*)alloc((size_t)G4 * EDIM * 2);      //   4 MB  [4H, E]
  bf16_t* Wh0T  = (bf16_t*)alloc((size_t)G4 * HID * 2);       //   8 MB  [4H, H]
  bf16_t* Wx1T  = (bf16_t*)alloc((size_t)G4 * HID * 2);       //   8 MB
  bf16_t* Wh1T  = (bf16_t*)alloc((size_t)G4 * HID * 2);       //   8 MB
  bf16_t* ebf   = (bf16_t*)alloc((size_t)ROWS * EDIM * 2);    //  16 MB  [B*S, E]
  float*  gates = (float*) alloc((size_t)ROWS * G4 * 4);      // 256 MB  [B*S, 4H] (reused per layer)
  bf16_t* ys0   = (bf16_t*)alloc((size_t)ROWS * HID * 2);     //  32 MB  layer-0 outputs
  float*  gbuf  = (float*) alloc((size_t)BATCH * G4 * 4);     //   1 MB  per-step gates
  float*  cst   = (float*) alloc((size_t)BATCH * HID * 4);
  float*  hst   = (float*) alloc((size_t)BATCH * HID * 4);
  bf16_t* hbf   = (bf16_t*)alloc((size_t)BATCH * HID * 2);
  float*  bs0   = (float*) alloc((size_t)G4 * 4);             // bx0+bh0
  float*  bs1   = (float*) alloc((size_t)G4 * 4);             // bx1+bh1
  (void)ws_size; (void)in_sizes; (void)n_in; (void)out_size;

  // --- one-time conversions / weight transposes ---
  { size_t n = (size_t)ROWS * INDIM;
    f32_to_bf16_k<<<(unsigned)((n + 255) / 256), 256, 0, stream>>>(x, xbf, n); }
  { size_t n = (size_t)INDIM * EDIM;
    w_transpose_bf16_k<<<(unsigned)((n + 255) / 256), 256, 0, stream>>>(W_enc, WencT, INDIM, EDIM); }
  { size_t n = (size_t)EDIM * G4;
    w_transpose_bf16_k<<<(unsigned)((n + 255) / 256), 256, 0, stream>>>(Wx0, Wx0T, EDIM, G4); }
  { size_t n = (size_t)HID * G4;
    w_transpose_bf16_k<<<(unsigned)((n + 255) / 256), 256, 0, stream>>>(Wh0, Wh0T, HID, G4);
    w_transpose_bf16_k<<<(unsigned)((n + 255) / 256), 256, 0, stream>>>(Wx1, Wx1T, HID, G4);
    w_transpose_bf16_k<<<(unsigned)((n + 255) / 256), 256, 0, stream>>>(Wh1, Wh1T, HID, G4); }
  vec_add_k<<<G4 / 256, 256, 0, stream>>>(bx0, bh0, bs0, G4);
  vec_add_k<<<G4 / 256, 256, 0, stream>>>(bx1, bh1, bs1, G4);

  // --- encoder: e = PReLU(x @ W_enc + b_enc), bf16 out ---
  wmma_gemm_bf16<<<dim3(ROWS / 16, EDIM / 256), 128, 0, stream>>>(
      xbf, INDIM, WencT, INDIM, b_enc, nullptr, 0,
      nullptr, ebf, EDIM, alpha, INDIM, 1);

  // --- layer 0: batched x-gates = e @ Wx0 + (bx0+bh0) ---
  wmma_gemm_bf16<<<dim3(ROWS / 16, G4 / 256), 128, 0, stream>>>(
      ebf, EDIM, Wx0T, EDIM, bs0, nullptr, 0,
      gates, nullptr, G4, nullptr, EDIM, 0);

  // --- layer 0 recurrence ---
  hipMemcpyAsync(cst, c0, (size_t)BATCH * HID * 4, hipMemcpyDeviceToDevice, stream);
  f32_to_bf16_k<<<(BATCH * HID) / 256, 256, 0, stream>>>(h0, hbf, (size_t)BATCH * HID);
  for (int t = 0; t < SEQ; ++t) {
    wmma_gemm_bf16<<<dim3(BATCH / 16, G4 / 256), 128, 0, stream>>>(
        hbf, HID, Wh0T, HID, nullptr, gates + (size_t)t * G4, (long long)SEQ * G4,
        gbuf, nullptr, G4, nullptr, HID, 0);
    lstm_update_k<<<(BATCH * HID) / 256, 256, 0, stream>>>(gbuf, cst, hst, hbf, ys0, t);
  }

  // --- layer 1: batched x-gates = ys0 @ Wx1 + (bx1+bh1)  (reuses gates buffer) ---
  wmma_gemm_bf16<<<dim3(ROWS / 16, G4 / 256), 128, 0, stream>>>(
      ys0, HID, Wx1T, HID, bs1, nullptr, 0,
      gates, nullptr, G4, nullptr, HID, 0);

  // --- layer 1 recurrence ---
  hipMemcpyAsync(cst, c0 + (size_t)BATCH * HID, (size_t)BATCH * HID * 4,
                 hipMemcpyDeviceToDevice, stream);
  f32_to_bf16_k<<<(BATCH * HID) / 256, 256, 0, stream>>>(h0 + (size_t)BATCH * HID, hbf,
                                                         (size_t)BATCH * HID);
  for (int t = 0; t < SEQ; ++t) {
    wmma_gemm_bf16<<<dim3(BATCH / 16, G4 / 256), 128, 0, stream>>>(
        hbf, HID, Wh1T, HID, nullptr, gates + (size_t)t * G4, (long long)SEQ * G4,
        gbuf, nullptr, G4, nullptr, HID, 0);
    lstm_update_k<<<(BATCH * HID) / 256, 256, 0, stream>>>(gbuf, cst, hst, hbf, nullptr, t);
  }

  // --- classifier: out = h_last @ W_cls + b_cls ---
  classifier_k<<<(BATCH * 10 + 255) / 256, 256, 0, stream>>>(hst, Wcls, bcls, out);
}
